// MoE_30167850287537
// MI455X (gfx1250) — matrix operand order repaired and checked by reading, not compiled, not run on hardware
//
#include <hip/hip_runtime.h>
#include <hip/hip_bf16.h>

typedef __attribute__((ext_vector_type(16))) __bf16 v16bf;
typedef __attribute__((ext_vector_type(8)))  float  v8f;

#define N_TOK 8192
#define C_DIM 1024
#define H_DIM 2048
#define E_NUM 8
#define R_DIM 16
#define TILE_M 64

union FragBF { v16bf v; uint4 q[2]; };

// ---------------- conversion / layout kernels ----------------

__global__ __launch_bounds__(256) void cvt_x_kernel(const float* __restrict__ src,
                                                    __bf16* __restrict__ dst, int n) {
  int i = blockIdx.x * 256 + threadIdx.x;
  if (i < n) dst[i] = (__bf16)src[i];
}

// W1: [E][C][H] fp32 -> W1t: [E][H][C] bf16 (B-matrix K-contiguous)
__global__ __launch_bounds__(256) void cvt_w1_kernel(const float* __restrict__ W1,
                                                     __bf16* __restrict__ W1t) {
  long i = (long)blockIdx.x * 256 + threadIdx.x;
  if (i >= (long)E_NUM * H_DIM * C_DIM) return;
  int c = (int)(i % C_DIM);
  long r = i / C_DIM;
  int n = (int)(r % H_DIM);
  int e = (int)(r / H_DIM);
  W1t[i] = (__bf16)W1[((size_t)e * C_DIM + c) * H_DIM + n];
}

// W2: [E][H][C] fp32 -> W2t: [E][C][H] bf16
__global__ __launch_bounds__(256) void cvt_w2_kernel(const float* __restrict__ W2,
                                                     __bf16* __restrict__ W2t) {
  long i = (long)blockIdx.x * 256 + threadIdx.x;
  if (i >= (long)E_NUM * C_DIM * H_DIM) return;
  int h = (int)(i % H_DIM);
  long r = i / H_DIM;
  int c = (int)(r % C_DIM);
  int e = (int)(r / C_DIM);
  W2t[i] = (__bf16)W2[((size_t)e * H_DIM + h) * C_DIM + c];
}

__global__ void zero_kernel(int* __restrict__ p, int n) {
  int i = blockIdx.x * blockDim.x + threadIdx.x;
  if (i < n) p[i] = 0;
}

// ---------------- gate: cosine router + softmax top-1 ----------------

__global__ __launch_bounds__(256) void gate_kernel(const float* __restrict__ x,
                                                   const float* __restrict__ Wr,
                                                   const float* __restrict__ wg,
                                                   float* __restrict__ score,
                                                   int* __restrict__ eidx,
                                                   int* __restrict__ counts) {
  int wave = threadIdx.x >> 5, lane = threadIdx.x & 31;
  int tok = blockIdx.x * 8 + wave;
  if (tok >= N_TOK) return;

  float acc[R_DIM];
#pragma unroll
  for (int r = 0; r < R_DIM; ++r) acc[r] = 0.f;
  const float* xrow = x + (size_t)tok * C_DIM;
  for (int c = lane; c < C_DIM; c += 32) {
    float xv = xrow[c];
#pragma unroll
    for (int r = 0; r < R_DIM; ++r) acc[r] = fmaf(xv, Wr[r * C_DIM + c], acc[r]);
  }
#pragma unroll
  for (int r = 0; r < R_DIM; ++r) {
#pragma unroll
    for (int off = 16; off >= 1; off >>= 1) acc[r] += __shfl_xor(acc[r], off, 32);
  }

  int e = lane & 7;
  float nrm = 0.f, dot = 0.f;
#pragma unroll
  for (int r = 0; r < R_DIM; ++r) {
    float w = wg[e * R_DIM + r];
    nrm = fmaf(w, w, nrm);
    dot = fmaf(w, acc[r], dot);
  }
  nrm = fmaxf(sqrtf(nrm), 1e-4f);
  float logit = dot / nrm;

  float m = logit;
#pragma unroll
  for (int off = 4; off >= 1; off >>= 1) m = fmaxf(m, __shfl_xor(m, off, 32));
  float s = expf(logit - m);
  float sum = s;
#pragma unroll
  for (int off = 4; off >= 1; off >>= 1) sum += __shfl_xor(sum, off, 32);

  unsigned long long bal = __ballot(logit == m);
  int sel = __ffsll((unsigned long long)(bal & 0xffull)) - 1;

  if (lane == 0) {
    score[tok] = 1.f / sum;
    eidx[tok] = sel;
    atomicAdd(&counts[sel], 1);
  }
}

// ---------------- expert bucketing (64-token tiles) ----------------

__global__ void prefix_kernel(const int* __restrict__ counts, int* __restrict__ offs,
                              int* __restrict__ cursor, int* __restrict__ tiles) {
  if (threadIdx.x == 0 && blockIdx.x == 0) {
    int o = 0, t = 0;
    for (int e = 0; e < E_NUM; ++e) {
      offs[e] = o;
      cursor[e] = o;
      tiles[e] = t;
      o += counts[e];
      t += (counts[e] + TILE_M - 1) / TILE_M;
    }
    offs[E_NUM] = o;
    tiles[E_NUM] = t;
  }
}

__global__ __launch_bounds__(256) void scatter_kernel(const int* __restrict__ eidx,
                                                      int* __restrict__ cursor,
                                                      int* __restrict__ order) {
  int t = blockIdx.x * 256 + threadIdx.x;
  if (t >= N_TOK) return;
  int e = eidx[t];
  int p = atomicAdd(&cursor[e], 1);
  order[p] = t;
}

// ---------------- fused expert FFN: 64-token tile, 4x2 register-blocked WMMA ----

__global__ __launch_bounds__(256) void ffn_kernel(
    const __bf16* __restrict__ xb, const __bf16* __restrict__ W1t,
    const float* __restrict__ b1, const __bf16* __restrict__ W2t,
    const float* __restrict__ b2, const float* __restrict__ score,
    const int* __restrict__ order, const int* __restrict__ offs,
    const int* __restrict__ tiles, __bf16* __restrict__ hws,
    float* __restrict__ out) {
  __shared__ __bf16 xA[TILE_M * C_DIM];  // 128 KB: x tile, row-major [M][K]
  __shared__ float ssc[TILE_M];
  __shared__ int srow[TILE_M];

  int total = tiles[E_NUM];
  int bid = blockIdx.x;
  if (bid >= total) return;
  int e = 0;
  while (e < E_NUM - 1 && bid >= tiles[e + 1]) ++e;
  int t = bid - tiles[e];
  int base = offs[e] + t * TILE_M;
  int cnt = offs[e + 1] - offs[e];
  long slotbase = (long)bid * TILE_M;  // h rows live at hws[slotbase + m]

  int tid = threadIdx.x;
  if (tid < TILE_M) {
    int valid = (t * TILE_M + tid) < cnt;
    int row = valid ? order[base + tid] : -1;
    srow[tid] = row;
    ssc[tid] = valid ? score[row] : 0.f;
  }
  __syncthreads();

  // --- stage x tile via async global->LDS (ASYNCcnt path); zero invalid rows ---
  for (int ch = tid; ch < TILE_M * (C_DIM / 8); ch += 256) {  // 16B chunks
    int m = ch >> 7;                 // row 0..63
    int kc = (ch & 127) << 3;        // element offset in row
    int row = srow[m];
    __bf16* ldst = xA + m * C_DIM + kc;
    if (row >= 0) {
      unsigned lds_addr = (unsigned)(size_t)ldst;
      const __bf16* g = xb + (size_t)row * C_DIM + kc;
      asm volatile("global_load_async_to_lds_b128 %0, %1, off"
                   :: "v"(lds_addr), "v"(g)
                   : "memory");
    } else {
      *(uint4*)ldst = make_uint4(0u, 0u, 0u, 0u);
    }
  }
  asm volatile("s_wait_asynccnt 0" ::: "memory");
  __syncthreads();

  int wave = tid >> 5, lane = tid & 31;
  int half = lane >> 4, nn = lane & 15;

  // ---- Phase 1: h = gelu(x @ W1[e] + b1[e]); K = C_DIM; 8 WMMA per K-step ----
  for (int ntp = wave; ntp < H_DIM / 32; ntp += 8) {
    int nb0 = ntp * 32;
    v8f acc0[4], acc1[4];
#pragma unroll
    for (int ms = 0; ms < 4; ++ms) { acc0[ms] = (v8f){}; acc1[ms] = (v8f){}; }
    const __bf16* bp0 = W1t + ((size_t)e * H_DIM + nb0 + nn) * C_DIM;
    const __bf16* bp1 = bp0 + (size_t)16 * C_DIM;
    for (int k0 = 0; k0 < C_DIM; k0 += 32) {
      FragBF b0, b1f;
      const __bf16* bk0 = bp0 + k0 + half * 16;
      const __bf16* bk1 = bp1 + k0 + half * 16;
      b0.q[0] = *(const uint4*)(bk0);
      b0.q[1] = *(const uint4*)(bk0 + 8);
      b1f.q[0] = *(const uint4*)(bk1);
      b1f.q[1] = *(const uint4*)(bk1 + 8);
#pragma unroll
      for (int ms = 0; ms < 4; ++ms) {
        FragBF a;
        const __bf16* ap = xA + (ms * 16 + nn) * C_DIM + k0 + half * 16;
        a.q[0] = *(const uint4*)(ap);
        a.q[1] = *(const uint4*)(ap + 8);
        acc0[ms] = __builtin_amdgcn_wmma_f32_16x16x32_bf16(false, a.v, false, b0.v,
                                                           (short)0, acc0[ms], false, false);
        acc1[ms] = __builtin_amdgcn_wmma_f32_16x16x32_bf16(false, a.v, false, b1f.v,
                                                           (short)0, acc1[ms], false, false);
      }
    }
#pragma unroll
    for (int j = 0; j < 2; ++j) {
      int nb = nb0 + j * 16;
      float bias = b1[e * H_DIM + nb + nn];
#pragma unroll
      for (int ms = 0; ms < 4; ++ms) {
#pragma unroll
        for (int r = 0; r < 8; ++r) {
          int m = ms * 16 + half * 8 + r;
          float v = (j ? acc1[ms][r] : acc0[ms][r]) + bias;
          v = 0.5f * v * (1.0f + erff(v * 0.70710678118654752f));  // exact gelu
          hws[(size_t)(slotbase + m) * H_DIM + nb + nn] = (__bf16)v;
        }
      }
    }
  }

  __threadfence();   // make h stores visible through L2
  __syncthreads();

  // ---- Phase 2: y = (h @ W2[e] + b2[e]) * score; K = H_DIM ----
  for (int ntp = wave; ntp < C_DIM / 32; ntp += 8) {
    int nb0 = ntp * 32;
    v8f acc0[4], acc1[4];
#pragma unroll
    for (int ms = 0; ms < 4; ++ms) { acc0[ms] = (v8f){}; acc1[ms] = (v8f){}; }
    const __bf16* bp0 = W2t + ((size_t)e * C_DIM + nb0 + nn) * H_DIM;
    const __bf16* bp1 = bp0 + (size_t)16 * H_DIM;
    for (int k0 = 0; k0 < H_DIM; k0 += 32) {
      FragBF b0, b1f;
      const __bf16* bk0 = bp0 + k0 + half * 16;
      const __bf16* bk1 = bp1 + k0 + half * 16;
      b0.q[0] = *(const uint4*)(bk0);
      b0.q[1] = *(const uint4*)(bk0 + 8);
      b1f.q[0] = *(const uint4*)(bk1);
      b1f.q[1] = *(const uint4*)(bk1 + 8);
#pragma unroll
      for (int ms = 0; ms < 4; ++ms) {
        FragBF a;
        const __bf16* ap = hws + (size_t)(slotbase + ms * 16 + nn) * H_DIM + k0 + half * 16;
        a.q[0] = *(const uint4*)(ap);
        a.q[1] = *(const uint4*)(ap + 8);
        acc0[ms] = __builtin_amdgcn_wmma_f32_16x16x32_bf16(false, a.v, false, b0.v,
                                                           (short)0, acc0[ms], false, false);
        acc1[ms] = __builtin_amdgcn_wmma_f32_16x16x32_bf16(false, a.v, false, b1f.v,
                                                           (short)0, acc1[ms], false, false);
      }
    }
#pragma unroll
    for (int j = 0; j < 2; ++j) {
      int nb = nb0 + j * 16;
      float bias = b2[e * C_DIM + nb + nn];
#pragma unroll
      for (int ms = 0; ms < 4; ++ms) {
#pragma unroll
        for (int r = 0; r < 8; ++r) {
          int m = ms * 16 + half * 8 + r;
          int row = srow[m];
          if (row >= 0)
            out[(size_t)row * C_DIM + nb + nn] =
                ((j ? acc1[ms][r] : acc0[ms][r]) + bias) * ssc[m];
        }
      }
    }
  }
}

// ---------------- launch ----------------

extern "C" void kernel_launch(void* const* d_in, const int* in_sizes, int n_in,
                              void* d_out, int out_size, void* d_ws, size_t ws_size,
                              hipStream_t stream) {
  const float* x  = (const float*)d_in[0];
  const float* Wr = (const float*)d_in[1];
  const float* wg = (const float*)d_in[2];
  const float* W1 = (const float*)d_in[3];
  const float* b1 = (const float*)d_in[4];
  const float* W2 = (const float*)d_in[5];
  const float* b2 = (const float*)d_in[6];
  float* out = (float*)d_out;

  char* ws = (char*)d_ws;
  float* score = (float*)(ws);             // 32 KB
  int* eidx    = (int*)(ws + 32768);       // 32 KB
  int* counts  = (int*)(ws + 65536);       // 64 B
  int* cursor  = (int*)(ws + 65600);       // 64 B
  int* offs    = (int*)(ws + 65664);       // 64 B
  int* tiles   = (int*)(ws + 65728);       // 64 B
  int* order   = (int*)(ws + 65792);       // 32 KB
  __bf16* xb  = (__bf16*)(ws + 131072);                                   // 16 MiB
  __bf16* W1t = (__bf16*)(ws + 131072 + ((size_t)16 << 20));              // 32 MiB
  __bf16* W2t = (__bf16*)(ws + 131072 + ((size_t)48 << 20));              // 32 MiB
  __bf16* hws = (__bf16*)(ws + 131072 + ((size_t)80 << 20));              // ~36 MiB

  int xn = N_TOK * C_DIM;
  cvt_x_kernel<<<(xn + 255) / 256, 256, 0, stream>>>(x, xb, xn);
  long wn = (long)E_NUM * H_DIM * C_DIM;
  cvt_w1_kernel<<<(int)((wn + 255) / 256), 256, 0, stream>>>(W1, W1t);
  cvt_w2_kernel<<<(int)((wn + 255) / 256), 256, 0, stream>>>(W2, W2t);

  zero_kernel<<<1, 64, 0, stream>>>(counts, 16);
  gate_kernel<<<N_TOK / 8, 256, 0, stream>>>(x, Wr, wg, score, eidx, counts);
  prefix_kernel<<<1, 1, 0, stream>>>(counts, offs, cursor, tiles);
  scatter_kernel<<<N_TOK / 256, 256, 0, stream>>>(eidx, cursor, order);

  ffn_kernel<<<N_TOK / TILE_M + E_NUM, 256, 0, stream>>>(
      xb, W1t, b1, W2t, b2, score, order, offs, tiles, hws, out);
}
